// WMSA_79680233276355
// MI455X (gfx1250) — matrix operand order repaired and checked by reading, not compiled, not run on hardware
//
#include <hip/hip_runtime.h>

// ---------------------------------------------------------------------------
// Fused W-MSA (Swin window attention) for gfx1250 / MI455X.
// One 256-thread (8 x wave32) block per 7x7 window; all intermediates in LDS;
// all GEMMs via V_WMMA_F32_16X16X32_F16 (f16 in, f32 accumulate).
// - every fragment load is 2 x 16B ds_load_b128 per lane
// - A fragments are loop-invariant per wave and hoisted; N-tile loops unrolled
// - V stored d-major (VT) by the QKV epilogue so P@V's B load is contiguous
// - output staged in LDS (f32) and written out channel-major (coalesced runs)
// ---------------------------------------------------------------------------

typedef __attribute__((ext_vector_type(16))) _Float16 v16h;
typedef __attribute__((ext_vector_type(4)))  _Float16 v4h;
typedef __attribute__((ext_vector_type(8)))  float    v8f;

#define MM   49      // tokens per window (padded to 64)
#define CC   96
#define HH   224
#define WWID 224
#define NW   1024    // windows per image (32 x 32)
#define BATCH 16

// LDS strides (elements). All f16 strides are multiples of 8 halves (16 B)
// so every 8-half fragment run is b128-aligned.
#define XS   104     // X tile / stack tile (64 rows x 96 cols)
#define WQS  104     // qkv weight, row-major N x K (288 rows x 96 cols)
#define QKS  200     // q,k activations token-major (64 rows: h*64+{q:0,k:32})
#define VTS  72      // V transposed: per head 32 rows(d) x 64 cols(token)
#define SS   68      // scores (f32), 64 rows
#define PS   72      // probabilities (f16), 64 rows
#define WOS  104     // out weight, row-major N x K (96 x 96)
#define YS   104     // staged output (f32), 64 rows x 96 cols (reuses WQ space)

// LDS byte offsets (16B aligned). ST overlays Xw (dead after QKV GEMM);
// Yf overlays WQ (dead after QKV GEMM; 64*104*4 = 26624 <= 59904).
#define OFF_X    0        // 64*104*2   = 13312   (also ST)
#define OFF_WQ   13312    // 288*104*2  = 59904   (also Yf)
#define OFF_QK   73216    // 64*200*2   = 25600
#define OFF_VT   98816    // 3*32*72*2  = 13824
#define OFF_S    112640   // 64*68*4    = 17408
#define OFF_P    130048   // 64*72*2    = 9216
#define OFF_WO   139264   // 96*104*2   = 19968
#define OFF_BI   159232   // 507*4 -> pad 2048
#define LDS_BYTES 161280  // <= 163840 -> 2 blocks / 320KB WGP

// --- WMMA fragment loaders (wave32; contiguous k = 16*(lane>>4)+e for A and B) -------

// A matrix 16x32 f16: lane r = row0+(lane&15); element e -> K kbase+e,
// kbase = k0 + 16*(lane>>4). Two 16B runs.
__device__ __forceinline__ v16h ldsA(const _Float16* base, int stride,
                                     int row0, int k0, int lane) {
  const int r = row0 + (lane & 15);
  const _Float16* p = base + r * stride + k0 + ((lane >> 4) << 4);
  v16h a;
#pragma unroll
  for (int e = 0; e < 8; ++e) a[e] = p[e];
#pragma unroll
  for (int e = 0; e < 8; ++e) a[8 + e] = p[8 + e];
  return a;
}

// B matrix 32x16 f16 from a transposed source (B[k][n] = src[n][k]):
// lane -> column n0+(lane&15); element e -> K = k0 + 16*(lane>>4) + e.
// Contiguous 32B per lane -> 2 x ds_load_b128.
__device__ __forceinline__ v16h ldsBT(const _Float16* base, int stride,
                                      int k0, int n0, int lane) {
  const int n  = n0 + (lane & 15);
  const _Float16* p = base + n * stride + k0 + ((lane >> 4) << 4);
  v16h b;
#pragma unroll
  for (int e = 0; e < 16; ++e) b[e] = p[e];
  return b;
}

__device__ __forceinline__ v8f wmma32(v16h a, v16h b, v8f c) {
  return __builtin_amdgcn_wmma_f32_16x16x32_f16(
      /*neg_a=*/false, a, /*neg_b=*/false, b,
      /*c_mod=*/(short)0, c, /*reuse_a=*/false, /*reuse_b=*/false);
}

// --- pack weights: pure f32 -> f16 cast, layout unchanged (row-major N x K)
__global__ void pack_weights(const float* __restrict__ qkv_w,
                             const float* __restrict__ out_w,
                             _Float16* __restrict__ Bq,
                             _Float16* __restrict__ Bo) {
  int i = blockIdx.x * blockDim.x + threadIdx.x;
  if (i < 288 * 96) Bq[i] = (_Float16)qkv_w[i];
  int j = i - 288 * 96;
  if (j >= 0 && j < 96 * 96) Bo[j] = (_Float16)out_w[j];
}

// --- fused window attention ------------------------------------------------
__global__ __launch_bounds__(256)
void wmsa_fused(const float* __restrict__ x,
                const float* __restrict__ qkv_b,
                const float* __restrict__ bias_table,
                const float* __restrict__ out_b,
                const _Float16* __restrict__ Bq,
                const _Float16* __restrict__ Bo,
                float* __restrict__ out) {
  extern __shared__ char smem[];
  _Float16* Xw = (_Float16*)(smem + OFF_X);
  _Float16* WQ = (_Float16*)(smem + OFF_WQ);
  _Float16* QK = (_Float16*)(smem + OFF_QK);
  _Float16* VT = (_Float16*)(smem + OFF_VT);
  float*    Sb = (float*)   (smem + OFF_S);
  _Float16* Pb = (_Float16*)(smem + OFF_P);
  _Float16* ST = (_Float16*)(smem + OFF_X);   // overlays Xw
  float*    Yf = (float*)   (smem + OFF_WQ);  // overlays WQ
  _Float16* WO = (_Float16*)(smem + OFF_WO);
  float*    BI = (float*)   (smem + OFF_BI);

  const int tid  = threadIdx.x;
  const int lane = tid & 31;
  const int wave = tid >> 5;

  const int wid = blockIdx.x;
  const int b   = wid >> 10;        // batch
  const int w   = wid & (NW - 1);
  const int row0 = (w >> 5) * 7;    // window pixel origin
  const int col0 = (w & 31) * 7;

  // per-wave tile coordinates (loop-invariant: tile stride == #waves == 8)
  const int mi    = wave & 3;            // M-tile
  const int nbase = wave >> 2;           // N-tile base (0 or 1)
  const int rbase = mi * 16 + ((lane >> 4) << 3);
  const int lc    = lane & 15;

  // ---- stage: bias table, weights (row-major N x K), X tile ----
  for (int i = tid; i < 507; i += 256) BI[i] = bias_table[i];

  for (int i = tid; i < 288 * 24; i += 256) {         // Wqkv: 288 rows x 24 v4h
    int r = i / 24, c = i % 24;
    *(v4h*)(WQ + r * WQS + c * 4) = *(const v4h*)(Bq + r * 96 + c * 4);
  }
  for (int i = tid; i < 96 * 24; i += 256) {          // Wout: 96 rows x 24 v4h
    int r = i / 24, c = i % 24;
    *(v4h*)(WO + r * WOS + c * 4) = *(const v4h*)(Bo + r * 96 + c * 4);
  }
  for (int i = tid; i < 96 * MM; i += 256) {          // gather 7x7 patch, all channels
    int c = i / MM, t = i % MM;
    int pr = row0 + t / 7, pc = col0 + t % 7;
    Xw[t * XS + c] = (_Float16)x[(((size_t)b * CC + c) * HH + pr) * WWID + pc];
  }
  for (int i = tid; i < 15 * 96; i += 256) {          // zero token rows 49..63
    int r = MM + i / 96, c = i % 96;
    Xw[r * XS + c] = (_Float16)0.f;
  }
  __syncthreads();

  // ---- GEMM1: QKV = X (64x96) @ Wqkv^T (96x288) + b ----
  // A fragments hoisted (same M-tile for all 9 N-tiles of this wave).
  {
    v16h a0 = ldsA(Xw, XS, mi * 16, 0,  lane);
    v16h a1 = ldsA(Xw, XS, mi * 16, 32, lane);
    v16h a2 = ldsA(Xw, XS, mi * 16, 64, lane);
#pragma unroll
    for (int it = 0; it < 9; ++it) {
      int ni = nbase + 2 * it;                 // 0..17
      v8f acc = {};
      acc = wmma32(a0, ldsBT(WQ, WQS, 0,  ni * 16, lane), acc);
      acc = wmma32(a1, ldsBT(WQ, WQS, 32, ni * 16, lane), acc);
      acc = wmma32(a2, ldsBT(WQ, WQS, 64, ni * 16, lane), acc);
      int cg  = ni * 16 + lc;                  // 0..287
      int h   = ni / 6;                        // head (16-col band uniform)
      int cwb = (ni - h * 6) * 16;             // within-head band base
      float bb = qkv_b[cg];
      if (cwb < 64) {                          // q or k: token-major
#pragma unroll
        for (int i2 = 0; i2 < 8; ++i2)
          QK[(rbase + i2) * QKS + h * 64 + cwb + lc] = (_Float16)(acc[i2] + bb);
      } else {                                 // v: store transposed (d-major)
#pragma unroll
        for (int i2 = 0; i2 < 8; ++i2)
          VT[(h * 32 + cwb - 64 + lc) * VTS + rbase + i2] = (_Float16)(acc[i2] + bb);
      }
    }
  }

  // ---- per-head attention ----
  const float inv_scale = 0.0180421963f;   // sqrt(3)/96  (attn /= C/sqrt(NH))
  for (int h = 0; h < 3; ++h) {
    __syncthreads();                       // QKV ready / previous head's P consumed
    const int qoff = h * 64, koff = qoff + 32;
    const _Float16* VTh = VT + h * 32 * VTS;

    // S = q @ k^T : 4x4 tiles, K=32; k^T load is contiguous
    {
      v16h a = ldsA(QK + qoff, QKS, mi * 16, 0, lane);
#pragma unroll
      for (int it = 0; it < 2; ++it) {
        int ni = nbase + 2 * it;             // covers 0..3 across waves
        v8f acc = {};
        acc = wmma32(a, ldsBT(QK + koff, QKS, 0, ni * 16, lane), acc);
        int cg = ni * 16 + lc;
        int cgd = cg / 7, cgm = cg - cgd * 7;
#pragma unroll
        for (int i2 = 0; i2 < 8; ++i2) {
          int rg = rbase + i2;
          float v;
          if (rg < MM && cg < MM) {
            int dh = rg / 7 - cgd + 6;
            int dw = rg % 7 - cgm + 6;
            v = acc[i2] * inv_scale + BI[(dh * 13 + dw) * 3 + h];
          } else {
            v = -1e30f;                      // mask padding -> exp() == 0
          }
          Sb[rg * SS + cg] = v;
        }
      }
    }
    __syncthreads();

    // softmax: 4 lanes per row (16 cols each), scores cached in registers
    {
      int r  = tid >> 2;                     // 0..63
      int q4 = tid & 3;
      _Float16* prow = Pb + r * PS + q4 * 16;
      if (r < MM) {
        const float* srow = Sb + r * SS + q4 * 16;
        float v[16];
#pragma unroll
        for (int j = 0; j < 16; ++j) v[j] = srow[j];
        float m = v[0];
#pragma unroll
        for (int j = 1; j < 16; ++j) m = fmaxf(m, v[j]);
        m = fmaxf(m, __shfl_xor(m, 1, 32));
        m = fmaxf(m, __shfl_xor(m, 2, 32));
        float s = 0.f;
#pragma unroll
        for (int j = 0; j < 16; ++j) { v[j] = __expf(v[j] - m); s += v[j]; }
        s += __shfl_xor(s, 1, 32);
        s += __shfl_xor(s, 2, 32);
        float inv = 1.f / s;
#pragma unroll
        for (int j = 0; j < 16; ++j) {
          int col = q4 * 16 + j;
          prow[j] = (col < MM) ? (_Float16)(v[j] * inv) : (_Float16)0.f;
        }
      } else {
#pragma unroll
        for (int j = 0; j < 16; ++j) prow[j] = (_Float16)0.f;
      }
    }
    __syncthreads();

    // O = P (64x64) @ V (64x32): B[k][n] = VT[h][n][k] -> contiguous
    {
      int ni = nbase;                         // 8 waves cover 4x2 tiles exactly
      v8f acc = {};
      acc = wmma32(ldsA(Pb, PS, mi * 16, 0,  lane),
                   ldsBT(VTh, VTS, 0,  ni * 16, lane), acc);
      acc = wmma32(ldsA(Pb, PS, mi * 16, 32, lane),
                   ldsBT(VTh, VTS, 32, ni * 16, lane), acc);
      int cg = ni * 16 + lc;
#pragma unroll
      for (int i2 = 0; i2 < 8; ++i2)
        ST[(rbase + i2) * XS + h * 32 + cg] = (_Float16)acc[i2];
    }
  }
  __syncthreads();

  // ---- out-proj: Y = stack (64x96) @ Wout^T (96x96) + b -> staged in LDS ----
  {
    v16h a0 = ldsA(ST, XS, mi * 16, 0,  lane);
    v16h a1 = ldsA(ST, XS, mi * 16, 32, lane);
    v16h a2 = ldsA(ST, XS, mi * 16, 64, lane);
#pragma unroll
    for (int it = 0; it < 3; ++it) {
      int ni = nbase + 2 * it;               // 0..5
      v8f acc = {};
      acc = wmma32(a0, ldsBT(WO, WOS, 0,  ni * 16, lane), acc);
      acc = wmma32(a1, ldsBT(WO, WOS, 32, ni * 16, lane), acc);
      acc = wmma32(a2, ldsBT(WO, WOS, 64, ni * 16, lane), acc);
      int cg = ni * 16 + lc;
      float ob = out_b[cg];
#pragma unroll
      for (int i2 = 0; i2 < 8; ++i2)
        Yf[(rbase + i2) * YS + cg] = acc[i2] + ob;
    }
  }
  __syncthreads();

  // ---- cooperative write-out: channel-major, contiguous 7-float pixel runs
  for (int i = tid; i < 96 * MM; i += 256) {
    int c = i / MM, t = i % MM;
    int pr = row0 + t / 7, pc = col0 + t % 7;
    out[(((size_t)b * CC + c) * HH + pr) * WWID + pc] = Yf[t * YS + c];
  }
}

extern "C" void kernel_launch(void* const* d_in, const int* in_sizes, int n_in,
                              void* d_out, int out_size, void* d_ws, size_t ws_size,
                              hipStream_t stream) {
  const float* x  = (const float*)d_in[0];
  const float* qw = (const float*)d_in[1];
  const float* qb = (const float*)d_in[2];
  const float* bt = (const float*)d_in[3];
  const float* ow = (const float*)d_in[4];
  const float* ob = (const float*)d_in[5];
  float* out = (float*)d_out;

  _Float16* Bq = (_Float16*)d_ws;                   // 288 x 96 f16 (55296 B)
  _Float16* Bo = (_Float16*)((char*)d_ws + 55296);  // 96 x 96  f16 (18432 B)

  pack_weights<<<(36864 + 255) / 256, 256, 0, stream>>>(qw, ow, Bq, Bo);
  wmsa_fused<<<BATCH * NW, 256, LDS_BYTES, stream>>>(x, qb, bt, ob, Bq, Bo, out);
}
